// TAttention_13563506720796
// MI455X (gfx1250) — matrix operand order repaired and checked by reading, not compiled, not run on hardware
//
#include <hip/hip_runtime.h>
#include <hip/hip_bf16.h>
#include <stdint.h>

#define B_  8
#define S_  2048
#define D_  1024
#define H_  16
#define DH_ 64
#define M_  (B_*S_)   // 16384 rows

typedef _Float16 h16;
typedef __attribute__((ext_vector_type(8)))  _Float16 v8h;
typedef __attribute__((ext_vector_type(16))) _Float16 v16h;
typedef __attribute__((ext_vector_type(8)))  float    v8f;
typedef __attribute__((ext_vector_type(4)))  uint32_t u32x4;
typedef __attribute__((ext_vector_type(8)))  uint32_t u32x8;

__device__ __forceinline__ v8f vzero8f(){
  v8f r;
#pragma unroll
  for (int i = 0; i < 8; ++i) r[i] = 0.0f;
  return r;
}

__device__ __forceinline__ v16h cat16(v8h a, v8h b){
  v16h r;
#pragma unroll
  for (int i = 0; i < 8; ++i) { r[i] = a[i]; r[i + 8] = b[i]; }
  return r;
}

// A-matrix fragment (16x32 f16): lane-half 0 holds K 0..7 & 16..23, half 1 K 8..15 & 24..31.
__device__ __forceinline__ v16h ldsA16(const h16* base){
  v8h lo = *(const v8h*)(base);
  v8h hh = *(const v8h*)(base + 16);
  return cat16(lo, hh);
}
// B-matrix fragment (32x16 f16): lane-half 0 holds K 0..7 & 16..23, half 1 K 8..15 & 24..31 of its column.
__device__ __forceinline__ v16h ldsB16(const h16* base){
  v8h lo = *(const v8h*)(base);
  v8h hh = *(const v8h*)(base + 16);
  return cat16(lo, hh);
}

__device__ __forceinline__ v8f wmma32(v16h a, v16h b, v8f c){
  return __builtin_amdgcn_wmma_f32_16x16x32_f16(false, a, false, b, (short)0, c, false, false);
}

__device__ __forceinline__ uint32_t lds_off_u32(const void* p){
  // generic LDS pointer: low 32 bits carry the LDS byte offset
  return (uint32_t)(uintptr_t)p;
}

// Per-lane async copy of 16B global -> LDS (ASYNCcnt tracked)
__device__ __forceinline__ void async_copy_b128(uint32_t lds_byte_off, const void* g){
  asm volatile("global_load_async_to_lds_b128 %0, %1, off"
               :: "v"(lds_byte_off), "v"((uint64_t)(uintptr_t)g)
               : "memory");
}
__device__ __forceinline__ void wait_async0(){
  asm volatile("s_wait_asynccnt 0x0" ::: "memory");
}

// TDM: one-instruction 2-D tile load global->LDS (TENSORcnt tracked).
// Tile tile1 rows x tile0 f16 elements, global row stride = stride0 elements.
// LDS rows padded: after every 16 DWORDs (32 halves) insert 4 DWORDs (8 halves)
// -> 40-half (80B) LDS row stride, conflict-free for the b128 fragment reads.
__device__ __forceinline__ void tdm_load_tile_2d(uint32_t lds_off, const void* gp,
                                                 uint32_t tile0, uint32_t tile1,
                                                 uint32_t stride0){
  uint64_t ga = (uint64_t)(uintptr_t)gp;
  u32x4 g0;
  g0[0] = 1u;                                               // count=1 (valid user D#)
  g0[1] = lds_off;                                          // lds_addr
  g0[2] = (uint32_t)ga;                                     // global_addr[31:0]
  g0[3] = ((uint32_t)(ga >> 32) & 0x01FFFFFFu) | (2u << 30);// global_addr[56:32] | type=2
  u32x8 g1;
  g1[0] = (1u << 16)        // data_size = 1 -> 2 bytes
        | (1u << 20)        // pad_enable
        | (3u << 22)        // pad_interval: 16 DWORDs
        | (3u << 25);       // pad_amount: 4 DWORDs
  g1[1] = (tile0 & 0xFFFFu) << 16;            // tensor_dim0[15:0]
  g1[2] = (tile1 & 0xFFFFu) << 16;            // tensor_dim0[31:16]=0 | tensor_dim1[15:0]
  g1[3] = (tile0 << 16);                      // tensor_dim1[31:16]=0 | tile_dim0
  g1[4] = tile1;                              // tile_dim1 (tile_dim2 = 0)
  g1[5] = stride0;                            // tensor_dim0_stride[31:0]
  g1[6] = 0u;                                 // stride hi | tensor_dim1_stride lo
  g1[7] = 0u;
  asm volatile("tensor_load_to_lds %0, %1" :: "s"(g0), "s"(g1) : "memory");
}

// ---------------------------------------------------------------------------
// fp32 -> fp16 conversion (weights)
// ---------------------------------------------------------------------------
__global__ void f32_to_f16_kernel(const float* __restrict__ in, h16* __restrict__ out, int n){
  int i = blockIdx.x * blockDim.x + threadIdx.x;
  if (i < n) out[i] = (h16)in[i];
}

// ---------------------------------------------------------------------------
// LayerNorm (optionally fused residual add): one 256-thread block per row.
// ---------------------------------------------------------------------------
template<bool ADD>
__global__ __launch_bounds__(256)
void ln_kernel(const float* __restrict__ x, const float* __restrict__ res,
               const float* __restrict__ g, const float* __restrict__ bta,
               float* __restrict__ outf, h16* __restrict__ outh){
  __shared__ float red[256];
  const int row = blockIdx.x;
  const int tid = threadIdx.x;
  const size_t rb = (size_t)row * D_;
  float v[4];
#pragma unroll
  for (int i = 0; i < 4; ++i){
    int c = tid * 4 + i;
    v[i] = x[rb + c];
    if (ADD) v[i] += res[rb + c];
  }
  float s = v[0] + v[1] + v[2] + v[3];
  red[tid] = s; __syncthreads();
  for (int o = 128; o > 0; o >>= 1){
    if (tid < o) red[tid] += red[tid + o];
    __syncthreads();
  }
  float mu = red[0] * (1.0f / D_);
  __syncthreads();
  float q = 0.f;
#pragma unroll
  for (int i = 0; i < 4; ++i){ float d = v[i] - mu; q += d * d; }
  red[tid] = q; __syncthreads();
  for (int o = 128; o > 0; o >>= 1){
    if (tid < o) red[tid] += red[tid + o];
    __syncthreads();
  }
  float rstd = rsqrtf(red[0] * (1.0f / D_) + 1e-5f);
#pragma unroll
  for (int i = 0; i < 4; ++i){
    int c = tid * 4 + i;
    float y = (v[i] - mu) * rstd * g[c] + bta[c];
    if (outf) outf[rb + c] = y;
    if (outh) outh[rb + c] = (h16)y;
  }
}

// ---------------------------------------------------------------------------
// WMMA GEMM: C[M,N] = A[M,K] @ W[N,K]^T (f16 row-major, K = 1024, N = 1024)
// Block tile 128x128, K-step 32, 8 wave32s (4x2), wave tile 32x64 (8 WMMAs).
// Staging: TDM tensor_load_to_lds, double buffered; wave 0 issues, TENSORcnt.
// ---------------------------------------------------------------------------
template<bool BIAS, bool RELU, bool RES, bool OUTF, bool OUTH>
__global__ __launch_bounds__(256)
void gemm_wmma_kernel(const h16* __restrict__ A, const h16* __restrict__ W,
                      const float* __restrict__ bias, const float* __restrict__ res,
                      float* __restrict__ outf, h16* __restrict__ outh){
  __shared__ h16 shA[2][128 * 40];   // padded rows: 40 halves = 80B
  __shared__ h16 shB[2][128 * 40];
  const int tid  = threadIdx.x;
  const int lane = tid & 31;
  const int lr   = lane & 15;
  const int hi   = lane >> 4;
  const int w    = tid >> 5;
  const int wm   = w & 3;        // 4 row strips of 32
  const int wn   = w >> 2;       // 2 col strips of 64
  const int m0   = blockIdx.y * 128;
  const int n0   = blockIdx.x * 128;

  v8f acc[2][4];
#pragma unroll
  for (int i = 0; i < 2; ++i)
#pragma unroll
    for (int j = 0; j < 4; ++j) acc[i][j] = vzero8f();

  const h16* Abase = A + (size_t)m0 * D_;
  const h16* Wbase = W + (size_t)n0 * D_;
  const uint32_t ldsA0 = lds_off_u32(&shA[0][0]);
  const uint32_t ldsB0 = lds_off_u32(&shB[0][0]);
  const uint32_t bufBytes = 128u * 40u * 2u;    // 10240
  const bool issuer = (tid < 32);               // wave 0 drives the TDM

  if (issuer){
    tdm_load_tile_2d(ldsA0, Abase, 32u, 128u, (uint32_t)D_);
    tdm_load_tile_2d(ldsB0, Wbase, 32u, 128u, (uint32_t)D_);
  }

  int buf = 0;
  for (int kk = 0; kk < D_; kk += 32){
    __syncthreads();   // all reads of the buffer we are about to refill are done
    if (issuer){
      if (kk + 32 < D_){
        uint32_t nb = (uint32_t)(buf ^ 1);
        tdm_load_tile_2d(ldsA0 + nb * bufBytes, Abase + kk + 32, 32u, 128u, (uint32_t)D_);
        tdm_load_tile_2d(ldsB0 + nb * bufBytes, Wbase + kk + 32, 32u, 128u, (uint32_t)D_);
        __builtin_amdgcn_s_wait_tensorcnt(2);   // in-order: current buffer's 2 loads done
      } else {
        __builtin_amdgcn_s_wait_tensorcnt(0);
      }
    }
    __syncthreads();   // current buffer visible to all waves

    const h16* sA = &shA[buf][0];
    const h16* sB = &shB[buf][0];
    v16h af[2], bf[4];
#pragma unroll
    for (int i = 0; i < 2; ++i)
      af[i] = ldsA16(sA + (wm * 32 + i * 16 + lr) * 40 + hi * 8);
#pragma unroll
    for (int j = 0; j < 4; ++j)
      bf[j] = ldsB16(sB + (wn * 64 + j * 16 + lr) * 40 + hi * 8);
#pragma unroll
    for (int i = 0; i < 2; ++i)
#pragma unroll
      for (int j = 0; j < 4; ++j)
        acc[i][j] = wmma32(af[i], bf[j], acc[i][j]);
    buf ^= 1;
  }

  // Epilogue. C/D layout: VGPR slot r, lane-half hi -> row r + 8*hi, col = lr.
#pragma unroll
  for (int i = 0; i < 2; ++i)
#pragma unroll
    for (int j = 0; j < 4; ++j)
#pragma unroll
      for (int r = 0; r < 8; ++r){
        int m = m0 + wm * 32 + i * 16 + r + hi * 8;
        int n = n0 + wn * 64 + j * 16 + lr;
        float vl = acc[i][j][r];
        if (BIAS) vl += bias[n];
        if (RELU) vl = fmaxf(vl, 0.0f);
        if (RES)  vl += res[(size_t)m * D_ + n];
        if (OUTF) outf[(size_t)m * D_ + n] = vl;
        if (OUTH) outh[(size_t)m * D_ + n] = (h16)vl;
      }
}

// ---------------------------------------------------------------------------
// Flash attention: block = (b, h, 128 q-rows), 8 waves, wave owns 16 q-rows.
// K tile staged via global_load_async_to_lds_b128 (ASYNCcnt); V transposed
// manually into LDS; QK^T and PV via WMMA; online softmax.
// ---------------------------------------------------------------------------
__global__ __launch_bounds__(256)
void flash_attn_kernel(const h16* __restrict__ Q, const h16* __restrict__ K,
                       const h16* __restrict__ V, float* __restrict__ O){
  __shared__ h16 shK [64 * 72];       // K tile, row stride 72 halves (144B)
  __shared__ h16 shVt[64 * 72];       // V^T tile: shVt[d][k]
  __shared__ h16 shP [8 * 16 * 72];   // per-wave P strip (16 x 64)
  const int tid  = threadIdx.x;
  const int lane = tid & 31, lr = lane & 15, hi = lane >> 4;
  const int w    = tid >> 5;
  const int qt = blockIdx.x, h = blockIdx.y, b = blockIdx.z;
  const size_t base = (size_t)b * S_ * D_ + (size_t)h * DH_;
  const int qrow0 = qt * 128 + w * 16;
  const uint32_t shKoff = lds_off_u32(shK);

  // Q strip fragments (A layout), 2 K-chunks of 32 over Dh=64
  v16h qf[2];
#pragma unroll
  for (int t = 0; t < 2; ++t){
    const h16* p = Q + base + (size_t)(qrow0 + lr) * D_ + t * 32 + hi * 8;
    qf[t] = cat16(*(const v8h*)p, *(const v8h*)(p + 16));
  }

  v8f o[4];
#pragma unroll
  for (int i = 0; i < 4; ++i) o[i] = vzero8f();
  float mrow[8], lrow[8];
#pragma unroll
  for (int r = 0; r < 8; ++r){ mrow[r] = -1e30f; lrow[r] = 0.0f; }

  for (int kt = 0; kt < S_; kt += 64){
    // stage K tile asynchronously; V tile loaded + transposed manually
#pragma unroll
    for (int i = 0; i < 2; ++i){
      int c = tid + 256 * i;
      int r = c >> 3, cc = c & 7;
      async_copy_b128(shKoff + (uint32_t)(r * 72 + cc * 8) * 2u,
                      K + base + (size_t)(kt + r) * D_ + cc * 8);
      v8h dv = *(const v8h*)(V + base + (size_t)(kt + r) * D_ + cc * 8);
#pragma unroll
      for (int j = 0; j < 8; ++j) shVt[(cc * 8 + j) * 72 + r] = dv[j];
      if (kt + 64 < S_)
        __builtin_prefetch(K + base + (size_t)(kt + 64 + r) * D_ + cc * 8, 0, 1);
    }
    wait_async0();
    __syncthreads();

    // scores strip: S[16 x 64] = Q_strip @ K_tile^T
    v8f sc[4];
#pragma unroll
    for (int nc = 0; nc < 4; ++nc) sc[nc] = vzero8f();
#pragma unroll
    for (int t = 0; t < 2; ++t)
#pragma unroll
      for (int nc = 0; nc < 4; ++nc){
        v16h kf = ldsB16(shK + (nc * 16 + lr) * 72 + t * 32 + hi * 8);
        sc[nc] = wmma32(qf[t], kf, sc[nc]);
      }

    // online softmax (scale = 1/sqrt(64) = 0.125)
    float mnew[8], alpha[8];
#pragma unroll
    for (int r = 0; r < 8; ++r){
      float tm = -1e30f;
#pragma unroll
      for (int nc = 0; nc < 4; ++nc) tm = fmaxf(tm, sc[nc][r]);
#pragma unroll
      for (int mask = 1; mask < 16; mask <<= 1)
        tm = fmaxf(tm, __shfl_xor(tm, mask, 16));
      tm *= 0.125f;
      float mn = fmaxf(mrow[r], tm);
      alpha[r] = __expf(mrow[r] - mn);
      mrow[r] = mn; mnew[r] = mn;
    }
#pragma unroll
    for (int r = 0; r < 8; ++r){
      float rs = 0.0f;
#pragma unroll
      for (int nc = 0; nc < 4; ++nc){
        float p = __expf(sc[nc][r] * 0.125f - mnew[r]);
        rs += p;
        shP[(w * 16 + r + hi * 8) * 72 + nc * 16 + lr] = (h16)p;
      }
#pragma unroll
      for (int mask = 1; mask < 16; mask <<= 1)
        rs += __shfl_xor(rs, mask, 16);
      lrow[r] = lrow[r] * alpha[r] + rs;
#pragma unroll
      for (int nd = 0; nd < 4; ++nd) o[nd][r] = o[nd][r] * alpha[r];
    }
    __syncthreads();

    // O_strip += P[16 x 64] @ V_tile[64 x 64]
#pragma unroll
    for (int t = 0; t < 2; ++t){
      const h16* pp = shP + (size_t)(w * 16 + lr) * 72 + t * 32 + hi * 8;
      v16h pf = cat16(*(const v8h*)pp, *(const v8h*)(pp + 16));
#pragma unroll
      for (int nd = 0; nd < 4; ++nd){
        v16h vf = ldsB16(shVt + (nd * 16 + lr) * 72 + t * 32 + hi * 8);
        o[nd] = wmma32(pf, vf, o[nd]);
      }
    }
    __syncthreads();
  }

  // normalize and store fp32 att [B,S,D]
#pragma unroll
  for (int nd = 0; nd < 4; ++nd)
#pragma unroll
    for (int r = 0; r < 8; ++r){
      int m = qrow0 + r + hi * 8;
      int n = nd * 16 + lr;
      O[base + (size_t)m * D_ + n] = o[nd][r] / lrow[r];
    }
}

// ---------------------------------------------------------------------------
extern "C" void kernel_launch(void* const* d_in, const int* in_sizes, int n_in,
                              void* d_out, int out_size, void* d_ws, size_t ws_size,
                              hipStream_t stream){
  const float* x    = (const float*)d_in[0];
  const float* wq   = (const float*)d_in[1];
  const float* wk   = (const float*)d_in[2];
  const float* wv   = (const float*)d_in[3];
  const float* ln1g = (const float*)d_in[4];
  const float* ln1b = (const float*)d_in[5];
  const float* ln2g = (const float*)d_in[6];
  const float* ln2b = (const float*)d_in[7];
  const float* w1   = (const float*)d_in[8];
  const float* b1   = (const float*)d_in[9];
  const float* w2   = (const float*)d_in[10];
  const float* b2   = (const float*)d_in[11];

  char* ws = (char*)d_ws;
  size_t off = 0;
  auto alloc = [&](size_t bytes)->char*{
    char* p = ws + off;
    off += (bytes + 255) & ~(size_t)255;
    return p;
  };
  float* xn_f  = (float*)alloc((size_t)M_ * D_ * 4);   // later reused as xt_f
  h16*   xn_h  = (h16*)  alloc((size_t)M_ * D_ * 2);
  h16*   q_h   = (h16*)  alloc((size_t)M_ * D_ * 2);   // later reused as h1_h
  h16*   k_h   = (h16*)  alloc((size_t)M_ * D_ * 2);   // later reused as xt_h
  h16*   v_h   = (h16*)  alloc((size_t)M_ * D_ * 2);
  float* att_f = (float*)alloc((size_t)M_ * D_ * 4);
  h16*   wq_h  = (h16*)  alloc((size_t)D_ * D_ * 2);
  h16*   wk_h  = (h16*)  alloc((size_t)D_ * D_ * 2);
  h16*   wv_h  = (h16*)  alloc((size_t)D_ * D_ * 2);
  h16*   w1_h  = (h16*)  alloc((size_t)D_ * D_ * 2);
  h16*   w2_h  = (h16*)  alloc((size_t)D_ * D_ * 2);
  float* xt_f  = xn_f;   // alias (safe: ln_kernel reads before writing)
  h16*   xt_h  = k_h;    // alias (k_h dead after flash attention)
  h16*   h1_h  = q_h;    // alias (q_h dead after flash attention)

  dim3 blk(256);
  const int NW = D_ * D_;
  f32_to_f16_kernel<<<dim3(NW / 256), blk, 0, stream>>>(wq, wq_h, NW);
  f32_to_f16_kernel<<<dim3(NW / 256), blk, 0, stream>>>(wk, wk_h, NW);
  f32_to_f16_kernel<<<dim3(NW / 256), blk, 0, stream>>>(wv, wv_h, NW);
  f32_to_f16_kernel<<<dim3(NW / 256), blk, 0, stream>>>(w1, w1_h, NW);
  f32_to_f16_kernel<<<dim3(NW / 256), blk, 0, stream>>>(w2, w2_h, NW);

  // LN1: xn (fp32 + fp16)
  ln_kernel<false><<<dim3(M_), blk, 0, stream>>>(x, nullptr, ln1g, ln1b, xn_f, xn_h);

  // Q/K/V projections (f16 out)
  dim3 gg(D_ / 128, M_ / 128);
  gemm_wmma_kernel<false,false,false,false,true><<<gg, blk, 0, stream>>>(xn_h, wq_h, nullptr, nullptr, nullptr, q_h);
  gemm_wmma_kernel<false,false,false,false,true><<<gg, blk, 0, stream>>>(xn_h, wk_h, nullptr, nullptr, nullptr, k_h);
  gemm_wmma_kernel<false,false,false,false,true><<<gg, blk, 0, stream>>>(xn_h, wv_h, nullptr, nullptr, nullptr, v_h);

  // fused flash attention -> att (fp32)
  dim3 ga(S_ / 128, H_, B_);
  flash_attn_kernel<<<ga, blk, 0, stream>>>(q_h, k_h, v_h, att_f);

  // LN2 over (xn + att) -> xt (fp32 + fp16); writes in place over xn_f
  ln_kernel<true><<<dim3(M_), blk, 0, stream>>>(xn_f, att_f, ln2g, ln2b, xt_f, xt_h);

  // FFN: h1 = relu(xt @ w1^T + b1); out = xt + (h1 @ w2^T + b2)
  gemm_wmma_kernel<true,true,false,false,true><<<gg, blk, 0, stream>>>(xt_h, w1_h, b1, nullptr, nullptr, h1_h);
  gemm_wmma_kernel<true,false,true,true,false><<<gg, blk, 0, stream>>>(h1_h, w2_h, b2, xt_f, (float*)d_out, nullptr);
}
